// SparseResidualBlock_25280177504760
// MI455X (gfx1250) — matrix operand order repaired and checked by reading, not compiled, not run on hardware
//
#include <hip/hip_runtime.h>
#include <hip/hip_bf16.h>

#define N_ACT 300000
#define C     128
#define KTAPS 9
#define EPSV  1e-4f
#define RED_BLOCKS 300
#define ROWS_PER_RED 1000

typedef __bf16 bf16_t;
typedef bf16_t v16bf __attribute__((ext_vector_type(16)));
typedef float  v8f   __attribute__((ext_vector_type(8)));

union AFrag { v16bf v; uint4 u[2]; };
union BFrag { v16bf v; uint4 u[2]; };

__device__ __forceinline__ unsigned short f32_to_bf16(float f) {
    unsigned u = __float_as_uint(f);
    u += 0x7FFFu + ((u >> 16) & 1u);        // round-to-nearest-even
    return (unsigned short)(u >> 16);
}

// ---------------------------------------------------------------------------
// Zero the 512B pad used for masked (idx < 0) gather rows
__global__ void zeropad_kernel(unsigned short* __restrict__ zpad)
{
    zpad[threadIdx.x] = 0;
}

// ---------------------------------------------------------------------------
// Convert f32 features -> bf16
__global__ __launch_bounds__(256) void tobf16_kernel(
    const float* __restrict__ x, unsigned short* __restrict__ xb, long n)
{
    long i = (long)blockIdx.x * blockDim.x + threadIdx.x;
    if (i < n) xb[i] = f32_to_bf16(x[i]);
}

// ---------------------------------------------------------------------------
// Pack W[k][cin][cout] (9x128x128 f32) into per-lane WMMA B-fragments (bf16).
// Fragment (k, kc, j) is the 32x16 block  B[kk][n] = W[k][kc*32+kk][j*16+n].
// Lane layout (16-bit B, K=32 x N=16): lanes 0-15 hold K {0..7,16..23} of column n=lane,
// lanes 16-31 hold K {8..15,24..31} of column n=lane-16; 16 bf16 contiguous per lane.
__global__ __launch_bounds__(256) void packw_kernel(
    const float* __restrict__ W, unsigned short* __restrict__ wfrag)
{
    int i = blockIdx.x * blockDim.x + threadIdx.x;   // 0 .. 9*16384-1
    if (i >= KTAPS * 16384) return;
    int e    = i & 15;
    int lane = (i >> 4) & 31;
    int j    = (i >> 9) & 7;
    int kc   = (i >> 12) & 3;
    int k    = i >> 14;
    int n    = lane & 15;
    int kk   = (lane >> 4) * 8 + (e >> 3) * 16 + (e & 7);
    float f  = W[(long)k * 16384 + (long)(kc * 32 + kk) * C + j * 16 + n];
    wfrag[i] = f32_to_bf16(f);
}

// ---------------------------------------------------------------------------
// Gathered sparse conv:  y[i][:] = sum_k  feats[nbr[i][k]][:] @ W[k]
// One wave per PAIR of 16-row tiles: 16 f32 WMMA accumulators (16x16) cover
// 2 x (16 rows x 128 out-channels).  B fragments staged in LDS per tap and
// software-pipelined; each B fragment feeds two independent WMMAs.
// Masked gathers are handled by redirecting the row base pointer to a zeroed
// pad (branchless, keeps A fragments in registers).
__global__ __launch_bounds__(256, 1) void conv_kernel(
    const unsigned short* __restrict__ feats,   // bf16 bits, N x 128
    const int*            __restrict__ nbr,     // N x 9
    const unsigned short* __restrict__ wfrag,   // packed bf16 weights
    const unsigned short* __restrict__ zpad,    // 256 zeroed bf16
    float*                __restrict__ yout,    // N x 128 f32
    int npairs)
{
    __shared__ unsigned short sw[16384];        // 32 KB = one tap's fragments

    const int tid  = threadIdx.x;
    const int lane = tid & 31;
    const int wave = tid >> 5;
    const int pair = blockIdx.x * 8 + wave;
    const bool active = (pair < npairs);
    const int p    = active ? pair : 0;
    const int t0   = p * 2;
    const int t1   = p * 2 + 1;

    const int n    = lane & 15;                 // A: row m, B/C: col n
    const int half = lane >> 4;

    v8f acc0[8], acc1[8];
    #pragma unroll
    for (int j = 0; j < 8; ++j)
        #pragma unroll
        for (int r = 0; r < 8; ++r) { acc0[j][r] = 0.f; acc1[j][r] = 0.f; }

    const long nb0 = (long)(t0 * 16 + n) * KTAPS;
    const long nb1 = (long)(t1 * 16 + n) * KTAPS;

    for (int k = 0; k < KTAPS; ++k) {
        __syncthreads();                         // LDS reuse fence
        {   // cooperative stage of tap-k weights: 2048 uint4 = 256 thr x 8
            const uint4* src = (const uint4*)(wfrag + (long)k * 16384);
            uint4* dst = (uint4*)sw;
            #pragma unroll
            for (int it = 0; it < 8; ++it)
                dst[it * 256 + tid] = src[it * 256 + tid];
        }
        __syncthreads();

        const int idx0 = nbr[nb0 + k];
        const int idx1 = nbr[nb1 + k];
        // Branchless masked gather: invalid neighbors read a zeroed pad.
        const unsigned short* b0 = (idx0 < 0) ? zpad : (feats + (long)idx0 * C);
        const unsigned short* b1 = (idx1 < 0) ? zpad : (feats + (long)idx1 * C);

        #pragma unroll 1                         // rolled: bound A liveness
        for (int kc = 0; kc < 4; ++kc) {
            // A fragment: 16-bit A 16x32; lane<16 -> K {0..7,16..23},
            // lane>=16 -> K {8..15,24..31} (relative to kc*32).
            AFrag A0, A1;
            const int c0 = kc * 32 + half * 8;
            A0.u[0] = *(const uint4*)(b0 + c0);
            A0.u[1] = *(const uint4*)(b0 + c0 + 16);
            A1.u[0] = *(const uint4*)(b1 + c0);
            A1.u[1] = *(const uint4*)(b1 + c0 + 16);

            // Software-pipelined B fragments from LDS
            const unsigned short* bbase = sw + kc * 4096 + lane * 16;
            BFrag Bc, Bn;
            Bc.u[0] = *(const uint4*)(bbase);
            Bc.u[1] = *(const uint4*)(bbase + 8);
            #pragma unroll
            for (int j = 0; j < 8; ++j) {
                if (j < 7) {
                    const unsigned short* bp = bbase + (j + 1) * 512;
                    Bn.u[0] = *(const uint4*)(bp);
                    Bn.u[1] = *(const uint4*)(bp + 8);
                }
                acc0[j] = __builtin_amdgcn_wmma_f32_16x16x32_bf16(
                    false, A0.v, false, Bc.v, (short)0, acc0[j], false, false);
                acc1[j] = __builtin_amdgcn_wmma_f32_16x16x32_bf16(
                    false, A1.v, false, Bc.v, (short)0, acc1[j], false, false);
                if (j < 7) Bc = Bn;
            }
        }
    }

    if (active) {
        // C/D layout: lanes 0-15 col n rows 0..7 ; lanes 16-31 col n rows 8..15
        #pragma unroll
        for (int j = 0; j < 8; ++j)
            #pragma unroll
            for (int r = 0; r < 8; ++r) {
                yout[(long)(t0 * 16 + half * 8 + r) * C + j * 16 + n] = acc0[j][r];
                yout[(long)(t1 * 16 + half * 8 + r) * C + j * 16 + n] = acc1[j][r];
            }
    }
}

// ---------------------------------------------------------------------------
// Deterministic two-pass batchnorm stats
__global__ __launch_bounds__(256) void reduce_kernel(
    const float* __restrict__ y, float* __restrict__ partial,
    int nrows, int rows_per_blk)
{
    const int tid = threadIdx.x;
    const int c   = tid & 127;
    const int sub = tid >> 7;
    int r0 = blockIdx.x * rows_per_blk;
    int r1 = r0 + rows_per_blk; if (r1 > nrows) r1 = nrows;
    float s = 0.f, ss = 0.f;
    for (int r = r0 + sub; r < r1; r += 2) {
        float v = y[(long)r * C + c];
        s += v; ss += v * v;
    }
    __shared__ float shs[256];
    __shared__ float shq[256];
    shs[tid] = s; shq[tid] = ss;
    __syncthreads();
    if (tid < 128) {
        partial[(long)blockIdx.x * 256 + tid]       = shs[tid] + shs[tid + 128];
        partial[(long)blockIdx.x * 256 + 128 + tid] = shq[tid] + shq[tid + 128];
    }
}

__global__ void stats_kernel(const float* __restrict__ partial, int nblk,
                             float* __restrict__ stats)   // [mean(128), rstd(128)]
{
    int c = threadIdx.x;
    float s = 0.f, ss = 0.f;
    for (int b = 0; b < nblk; ++b) {
        s  += partial[(long)b * 256 + c];
        ss += partial[(long)b * 256 + 128 + c];
    }
    float mean = s / (float)N_ACT;
    float var  = ss / (float)N_ACT - mean * mean;
    stats[c]       = mean;
    stats[128 + c] = rsqrtf(var + EPSV);
}

// ---------------------------------------------------------------------------
// h = relu(bn(y)) -> bf16 (input for conv2)
__global__ __launch_bounds__(256) void bnrelu_kernel(
    const float* __restrict__ y, const float* __restrict__ stats,
    const float* __restrict__ gamma, const float* __restrict__ beta,
    unsigned short* __restrict__ hb, long nelem)
{
    long i = (long)blockIdx.x * blockDim.x + threadIdx.x;
    if (i >= nelem) return;
    int c = (int)(i & (C - 1));
    float v = (y[i] - stats[c]) * stats[128 + c] * gamma[c] + beta[c];
    v = v > 0.f ? v : 0.f;
    hb[i] = f32_to_bf16(v);
}

// out = relu(bn(y2) + x)
__global__ __launch_bounds__(256) void final_kernel(
    const float* __restrict__ y, const float* __restrict__ stats,
    const float* __restrict__ gamma, const float* __restrict__ beta,
    const float* __restrict__ x, float* __restrict__ out, long nelem)
{
    long i = (long)blockIdx.x * blockDim.x + threadIdx.x;
    if (i >= nelem) return;
    int c = (int)(i & (C - 1));
    float v = (y[i] - stats[c]) * stats[128 + c] * gamma[c] + beta[c] + x[i];
    out[i] = v > 0.f ? v : 0.f;
}

// ---------------------------------------------------------------------------
extern "C" void kernel_launch(void* const* d_in, const int* in_sizes, int n_in,
                              void* d_out, int out_size, void* d_ws, size_t ws_size,
                              hipStream_t stream)
{
    const float* x   = (const float*)d_in[0];
    const int*   nbr = (const int*)  d_in[1];
    const float* W1  = (const float*)d_in[2];
    const float* g1  = (const float*)d_in[3];
    const float* b1  = (const float*)d_in[4];
    const float* W2  = (const float*)d_in[5];
    const float* g2  = (const float*)d_in[6];
    const float* b2  = (const float*)d_in[7];
    float* out = (float*)d_out;

    const long NC = (long)N_ACT * C;
    char* ws = (char*)d_ws;
    size_t off = 0;
    auto alloc = [&](size_t bytes) {
        char* p = ws + off;
        off = (off + bytes + 255) & ~(size_t)255;
        return p;
    };
    unsigned short* xb  = (unsigned short*)alloc(NC * 2);   // bf16 feats (reused as hb)
    float*          y   = (float*)         alloc(NC * 4);   // conv output (reused)
    unsigned short* wf1 = (unsigned short*)alloc((size_t)KTAPS * 16384 * 2);
    unsigned short* wf2 = (unsigned short*)alloc((size_t)KTAPS * 16384 * 2);
    float*          par = (float*)         alloc((size_t)RED_BLOCKS * 256 * 4);
    float*          st  = (float*)         alloc(256 * 4);
    unsigned short* zp  = (unsigned short*)alloc(256 * 2);  // zero pad for masked rows

    const int npairs  = N_ACT / 32;                        // 9375 tile pairs
    const int cblocks = (npairs + 7) / 8;
    const int eblocks = (int)((NC + 255) / 256);
    const int wblocks = (KTAPS * 16384 + 255) / 256;

    zeropad_kernel<<<1, 256, 0, stream>>>(zp);
    tobf16_kernel<<<eblocks, 256, 0, stream>>>(x, xb, NC);
    packw_kernel <<<wblocks, 256, 0, stream>>>(W1, wf1);
    packw_kernel <<<wblocks, 256, 0, stream>>>(W2, wf2);

    // layer 1
    conv_kernel  <<<cblocks, 256, 0, stream>>>(xb, nbr, wf1, zp, y, npairs);
    reduce_kernel<<<RED_BLOCKS, 256, 0, stream>>>(y, par, N_ACT, ROWS_PER_RED);
    stats_kernel <<<1, 128, 0, stream>>>(par, RED_BLOCKS, st);
    bnrelu_kernel<<<eblocks, 256, 0, stream>>>(y, st, g1, b1, xb, NC);

    // layer 2
    conv_kernel  <<<cblocks, 256, 0, stream>>>(xb, nbr, wf2, zp, y, npairs);
    reduce_kernel<<<RED_BLOCKS, 256, 0, stream>>>(y, par, N_ACT, ROWS_PER_RED);
    stats_kernel <<<1, 128, 0, stream>>>(par, RED_BLOCKS, st);
    final_kernel <<<eblocks, 256, 0, stream>>>(y, st, g2, b2, x, out, NC);
}
